// DGWA_38311108280838
// MI455X (gfx1250) — compile-verified, matrix operand order checked by reading, not run on hardware
//
#include <hip/hip_runtime.h>
#include <hip/hip_bf16.h>

// ---------------------------------------------------------------------------
// Deformable shifted-window attention, fused per-window, CDNA5 (gfx1250).
// One workgroup (256 threads = 8 wave32) per 8x8 window; all intermediates in
// LDS (~280KB of the 320KB WGP LDS); all GEMMs via v_wmma_f32_16x16x32_f16.
// Neighbor-window tiles are fetched with async global->LDS loads when the
// toolchain exposes them (ASYNCcnt path), else via vector loads.
// ---------------------------------------------------------------------------

typedef __attribute__((ext_vector_type(16))) _Float16 v16h;
typedef __attribute__((ext_vector_type(8)))  _Float16 v8h;
typedef __attribute__((ext_vector_type(4)))  _Float16 v4h;
typedef __attribute__((ext_vector_type(8)))  float    v8f;
typedef __attribute__((ext_vector_type(4)))  int      v4i;

#define WSZ    8      // window size
#define CCH    128    // channels
#define NH     4      // heads
#define HD     32     // head dim
#define P2     64     // tokens per window
#define GW     32     // windows per image dim
#define IMG    256
#define SHF    4      // shift

// ---- LDS layout (bytes) ----------------------------------------------------
#define OFF_XBUF   0u        // f32 [64][132]  own window tile (shifted x)
#define OFF_XD     33792u    // f32 [64][132]  inter-window sampled tile
#define OFF_XQ16   67584u    // f16 [64][136]  q-GEMM input
#define OFF_XKV16  84992u    // f16 [64][136]  kv-GEMM input (xd2)
#define OFF_QBUF   102400u   // f16 [64][136]  q output, token-major
#define OFF_KBUF   119808u   // f16 [64][136]  k output, token-major
#define OFF_VT     137216u   // f16 [128][72]  v output, channel-major (transposed)
#define OFF_SIM    155648u   // f32 [4][64][68] attention logits
#define OFF_PROBS  225280u   // f16 [4][64][72] softmax probs (A-fragments)
#define OFF_AOUT   262144u   // f16 [64][136]  attention output
#define OFF_OFFI   279552u   // f32 [64][2]    intra-window offsets
#define OFF_PART   280064u   // f32 [64][2]    oi reduction partials
#define OFF_OI     280576u   // f32 [2]        per-window offset
#define SMEM_BYTES 280592u
// Async staging for 4 corner tiles: f32 [4][64][128] = 131072B overlays the
// qbuf/kbuf/vT/sim regions (all dead during Phase C).
#define OFF_STAGE  OFF_QBUF

// ---- async global->LDS (guarded: falls back to vector loads) ---------------
#if __has_builtin(__builtin_amdgcn_global_load_async_to_lds_b128)
#define HAVE_ASYNC_LDS 1
__device__ inline void async_ld_b128(const float* gp, float* lp) {
  // Builtin signature (from hipcc diagnostic): param0 = v4i* in addrspace(1).
  __builtin_amdgcn_global_load_async_to_lds_b128(
      (__attribute__((address_space(1))) v4i*)gp,
      (__attribute__((address_space(3))) v4i*)lp, 0, 0);
}
__device__ inline void wait_async0() {
#if __has_builtin(__builtin_amdgcn_s_wait_asynccnt)
  __builtin_amdgcn_s_wait_asynccnt(0);
#else
  asm volatile("s_wait_asynccnt 0x0" ::: "memory");
#endif
}
#else
#define HAVE_ASYNC_LDS 0
#endif

// ---- WMMA helpers ----------------------------------------------------------
// 16-bit A (16x32 MxK) layout per ISA 7.12.2: lane L -> M = L%16,
// half hh -> K = 16*(hh/8) + 8*(L/16) + (hh%8).  B (32x16 KxN) mirrored with
// N = L%16.  Both reduce to: per lane, two contiguous 16B chunks 32B apart.
__device__ inline v16h load_frag16(const _Float16* base, int strideH,
                                   int row0, int k0, int lane) {
  int r  = row0 + (lane & 15);
  int kb = k0 + ((lane >> 4) << 3);
  const _Float16* p = base + r * strideH + kb;
  union { v16h v; v8h h[2]; } u;
  u.h[0] = *(const v8h*)(p);
  u.h[1] = *(const v8h*)(p + 16);
  return u.v;
}

// Weight B-fragments pre-packed as [tile][lane][hh]: 32 contiguous bytes/lane.
__device__ inline v16h load_fragw(const _Float16* pack, int tile, int lane) {
  return *(const v16h*)(pack + (size_t)tile * 512 + lane * 16);
}

__device__ inline v8f wmma16(v16h a, v16h b, v8f c) {
  return __builtin_amdgcn_wmma_f32_16x16x32_f16(false, a, false, b,
                                                (short)0, c, false, false);
}

__device__ inline float4 fma4(float4 a, float s, float4 acc) {
  acc.x += a.x * s; acc.y += a.y * s; acc.z += a.z * s; acc.w += a.w * s;
  return acc;
}

// ---------------------------------------------------------------------------
// Pre-pack kernel: convert f32 row-major weights [K][N] into f16 WMMA
// B-fragment order: [ntile][kstep][lane][hh].
// ---------------------------------------------------------------------------
__global__ __launch_bounds__(256)
void dgwa_pack(const float* __restrict__ qw, const float* __restrict__ kvw,
               const float* __restrict__ ow,
               _Float16* __restrict__ qp, _Float16* __restrict__ kvp,
               _Float16* __restrict__ op) {
  int idx = blockIdx.x * 256 + threadIdx.x;
  if (idx >= 65536) return;
  const float* src; _Float16* dst; int rem, ncols;
  if (idx < 16384)      { src = qw;  dst = qp;  rem = idx;          ncols = 128; }
  else if (idx < 49152) { src = kvw; dst = kvp; rem = idx - 16384;  ncols = 256; }
  else                  { src = ow;  dst = op;  rem = idx - 49152;  ncols = 128; }
  int nt   = rem >> 11;           // ntile (4 ksteps * 512 elems each)
  int ks   = (rem >> 9) & 3;      // kstep
  int e    = rem & 511;
  int lane = e >> 4;
  int hh   = e & 15;
  int K = ks * 32 + ((hh >> 3) << 4) + ((lane >> 4) << 3) + (hh & 7);
  int N = nt * 16 + (lane & 15);
  dst[rem] = (_Float16)src[(size_t)K * ncols + N];
}

// ---------------------------------------------------------------------------
// Main fused kernel: one block per (batch, window).
// ---------------------------------------------------------------------------
__global__ __launch_bounds__(256)
void dgwa_main(const float* __restrict__ x, const float* __restrict__ rpp,
               const float* __restrict__ wi_w, const float* __restrict__ wi_b,
               const float* __restrict__ w1_w, const float* __restrict__ w1_b,
               const float* __restrict__ w2_w, const float* __restrict__ w2_b,
               const float* __restrict__ q_b,  const float* __restrict__ kv_b,
               const float* __restrict__ o_b,
               const _Float16* __restrict__ qpack,
               const _Float16* __restrict__ kvpack,
               const _Float16* __restrict__ opack,
               float* __restrict__ out) {
  extern __shared__ char smem[];
  float*    xbuf  = (float*)   (smem + OFF_XBUF);   // stride 132
  float*    xd    = (float*)   (smem + OFF_XD);     // stride 132
  _Float16* xq16  = (_Float16*)(smem + OFF_XQ16);   // stride 136
  _Float16* xkv16 = (_Float16*)(smem + OFF_XKV16);  // stride 136
  _Float16* qbuf  = (_Float16*)(smem + OFF_QBUF);   // stride 136
  _Float16* kbuf  = (_Float16*)(smem + OFF_KBUF);   // stride 136
  _Float16* vT    = (_Float16*)(smem + OFF_VT);     // stride 72
  float*    sim   = (float*)   (smem + OFF_SIM);    // [4][64][68]
  _Float16* probs = (_Float16*)(smem + OFF_PROBS);  // [4][64][72]
  _Float16* aout  = (_Float16*)(smem + OFF_AOUT);   // stride 136
  float*    offi  = (float*)   (smem + OFF_OFFI);
  float*    part  = (float*)   (smem + OFF_PART);
  float*    oiS   = (float*)   (smem + OFF_OI);

  const int tid  = threadIdx.x;
  const int lane = tid & 31;
  const int wv   = tid >> 5;
  const int bb   = blockIdx.x >> 10;
  const int win  = blockIdx.x & 1023;
  const int wy   = win >> 5;
  const int wx   = win & 31;
  const float* xb = x + (size_t)bb * IMG * IMG * CCH;

  // ---- Phase A: load own (shifted) window tile; keep f32 + f16 copies -----
#if HAVE_ASYNC_LDS
  #pragma unroll
  for (int k = 0; k < 8; ++k) {
    int idx = tid + 256 * k;              // 2048 float4's
    int t  = idx >> 5, c4 = idx & 31;
    int i  = t >> 3,  j  = t & 7;
    int row = (wy * 8 + i + SHF) & 255;
    int col = (wx * 8 + j + SHF) & 255;
    async_ld_b128(xb + ((size_t)row * IMG + col) * CCH + c4 * 4,
                  xbuf + t * 132 + c4 * 4);
  }
  wait_async0();
  __syncthreads();
  #pragma unroll
  for (int k = 0; k < 8; ++k) {
    int idx = tid + 256 * k;
    int t = idx >> 5, c4 = idx & 31;
    float4 v = *(const float4*)(xbuf + t * 132 + c4 * 4);
    v4h hv; hv[0] = (_Float16)v.x; hv[1] = (_Float16)v.y;
            hv[2] = (_Float16)v.z; hv[3] = (_Float16)v.w;
    *(v4h*)(xq16 + t * 136 + c4 * 4) = hv;
  }
#else
  #pragma unroll
  for (int k = 0; k < 8; ++k) {
    int idx = tid + 256 * k;              // 2048 float4's
    int t  = idx >> 5, c4 = idx & 31;
    int i  = t >> 3,  j  = t & 7;
    int row = (wy * 8 + i + SHF) & 255;
    int col = (wx * 8 + j + SHF) & 255;
    float4 v = *(const float4*)(xb + ((size_t)row * IMG + col) * CCH + c4 * 4);
    *(float4*)(xbuf + t * 132 + c4 * 4) = v;
    v4h hv; hv[0] = (_Float16)v.x; hv[1] = (_Float16)v.y;
            hv[2] = (_Float16)v.z; hv[3] = (_Float16)v.w;
    *(v4h*)(xq16 + t * 136 + c4 * 4) = hv;
  }
  __syncthreads();
#endif

  // ---- Phase B: offset MLPs (per-token wi, per-window w1->w2) -------------
  if (tid < 64) {
    const float* xr = xbuf + tid * 132;
    float s0 = wi_b[0], s1 = wi_b[1], a0 = w1_b[0], a1 = w1_b[1];
    for (int c = 0; c < CCH; ++c) {
      float xv = xr[c];
      s0 += xv * wi_w[2 * c];     s1 += xv * wi_w[2 * c + 1];
      a0 += xv * w1_w[2 * c];     a1 += xv * w1_w[2 * c + 1];
    }
    offi[2 * tid] = s0; offi[2 * tid + 1] = s1;
    part[2 * tid]     = a0 * w2_w[2 * (2 * tid)]     + a1 * w2_w[2 * (2 * tid + 1)];
    part[2 * tid + 1] = a0 * w2_w[2 * (2 * tid) + 1] + a1 * w2_w[2 * (2 * tid + 1) + 1];
  }
  __syncthreads();
  if (tid < 2) {
    float s = w2_b[tid];
    for (int t = 0; t < 64; ++t) s += part[2 * t + tid];
    oiS[tid] = s;
  }
  __syncthreads();

  // ---- Phase C: inter-window bilinear sample = 4-neighbor tile blend ------
  {
    float sx = (float)wx + oiS[0];
    float sy = (float)wy + oiS[1];
    float fx = floorf(sx), fy = floorf(sy);
    int ix0 = (int)fx, iy0 = (int)fy;
    float u = sx - fx, v = sy - fy;
    float cw[4] = { (1.f - u) * (1.f - v), u * (1.f - v), (1.f - u) * v, u * v };
    int   cx[4] = { ix0, ix0 + 1, ix0,     ix0 + 1 };
    int   cy[4] = { iy0, iy0,     iy0 + 1, iy0 + 1 };
    bool  ok[4];
    #pragma unroll
    for (int cn = 0; cn < 4; ++cn)
      ok[cn] = (cx[cn] >= 0) && (cx[cn] < GW) && (cy[cn] >= 0) && (cy[cn] < GW);

#if HAVE_ASYNC_LDS
    float* stage = (float*)(smem + OFF_STAGE);   // [4][64][128] f32
    #pragma unroll
    for (int cn = 0; cn < 4; ++cn) {
      if (ok[cn]) {                               // block-uniform branch
        #pragma unroll
        for (int k = 0; k < 8; ++k) {
          int idx = tid + 256 * k;
          int t = idx >> 5, c4 = idx & 31;
          int i = t >> 3,  j  = t & 7;
          int row = (cy[cn] * 8 + i + SHF) & 255;
          int col = (cx[cn] * 8 + j + SHF) & 255;
          async_ld_b128(xb + ((size_t)row * IMG + col) * CCH + c4 * 4,
                        stage + cn * 8192 + t * 128 + c4 * 4);
        }
      }
    }
    wait_async0();
    __syncthreads();
    #pragma unroll
    for (int k = 0; k < 8; ++k) {
      int idx = tid + 256 * k;
      int t = idx >> 5, c4 = idx & 31;
      float4 acc = make_float4(0.f, 0.f, 0.f, 0.f);
      #pragma unroll
      for (int cn = 0; cn < 4; ++cn) {
        if (ok[cn]) {
          float4 g = *(const float4*)(stage + cn * 8192 + t * 128 + c4 * 4);
          acc = fma4(g, cw[cn], acc);
        }
      }
      *(float4*)(xd + t * 132 + c4 * 4) = acc;
    }
#else
    float4 acc[8];
    #pragma unroll
    for (int k = 0; k < 8; ++k) acc[k] = make_float4(0.f, 0.f, 0.f, 0.f);
    #pragma unroll
    for (int cn = 0; cn < 4; ++cn) {
      if (ok[cn]) {                               // block-uniform branch
        #pragma unroll
        for (int k = 0; k < 8; ++k) {
          int idx = tid + 256 * k;
          int t = idx >> 5, c4 = idx & 31;
          int i = t >> 3,  j  = t & 7;
          int row = (cy[cn] * 8 + i + SHF) & 255;
          int col = (cx[cn] * 8 + j + SHF) & 255;
          float4 g = *(const float4*)(xb + ((size_t)row * IMG + col) * CCH + c4 * 4);
          acc[k] = fma4(g, cw[cn], acc[k]);
        }
      }
    }
    #pragma unroll
    for (int k = 0; k < 8; ++k) {
      int idx = tid + 256 * k;
      int t = idx >> 5, c4 = idx & 31;
      *(float4*)(xd + t * 132 + c4 * 4) = acc[k];
    }
#endif
  }
  __syncthreads();

  // ---- Phase D: intra-window bilinear gather -> xkv16 ---------------------
  {
    int t  = tid >> 2;          // token
    int cg = tid & 3;           // 32-channel group
    int i = t >> 3, j = t & 7;
    float px = (float)j + offi[2 * t];
    float py = (float)i + offi[2 * t + 1];
    float fx = floorf(px), fy = floorf(py);
    int jx0 = (int)fx, iy0 = (int)fy;
    float u = px - fx, v = py - fy;
    bool vx0 = (jx0 >= 0) & (jx0 < 8),  vx1 = (jx0 + 1 >= 0) & (jx0 + 1 < 8);
    bool vy0 = (iy0 >= 0) & (iy0 < 8),  vy1 = (iy0 + 1 >= 0) & (iy0 + 1 < 8);
    float a00 = (vx0 && vy0) ? (1.f - u) * (1.f - v) : 0.f;
    float a10 = (vx1 && vy0) ? u * (1.f - v)         : 0.f;
    float a01 = (vx0 && vy1) ? (1.f - u) * v         : 0.f;
    float a11 = (vx1 && vy1) ? u * v                 : 0.f;
    int jc0 = min(max(jx0, 0), 7), jc1 = min(max(jx0 + 1, 0), 7);
    int ic0 = min(max(iy0, 0), 7), ic1 = min(max(iy0 + 1, 0), 7);
    int t00 = ic0 * 8 + jc0, t10 = ic0 * 8 + jc1;
    int t01 = ic1 * 8 + jc0, t11 = ic1 * 8 + jc1;
    #pragma unroll
    for (int k = 0; k < 8; ++k) {
      int c4 = cg * 8 + k;
      float4 r00 = *(const float4*)(xd + t00 * 132 + c4 * 4);
      float4 r10 = *(const float4*)(xd + t10 * 132 + c4 * 4);
      float4 r01 = *(const float4*)(xd + t01 * 132 + c4 * 4);
      float4 r11 = *(const float4*)(xd + t11 * 132 + c4 * 4);
      float4 r = make_float4(0.f, 0.f, 0.f, 0.f);
      r = fma4(r00, a00, r); r = fma4(r10, a10, r);
      r = fma4(r01, a01, r); r = fma4(r11, a11, r);
      v4h hv; hv[0] = (_Float16)r.x; hv[1] = (_Float16)r.y;
              hv[2] = (_Float16)r.z; hv[3] = (_Float16)r.w;
      *(v4h*)(xkv16 + t * 136 + c4 * 4) = hv;
    }
  }
  __syncthreads();

  // ---- Phase E: kv = xkv16 @ kv_w + kv_b  (64x128 @ 128x256) --------------
  #pragma unroll
  for (int rep = 0; rep < 8; ++rep) {
    int tile = wv + rep * 8;            // 0..63
    int mt = tile >> 4, nt = tile & 15;
    int n = nt * 16 + (lane & 15);
    float bv = kv_b[n];
    v8f c;
    #pragma unroll
    for (int r = 0; r < 8; ++r) c[r] = bv;
    #pragma unroll
    for (int ks = 0; ks < 4; ++ks) {
      v16h a = load_frag16(xkv16, 136, mt * 16, ks * 32, lane);
      v16h b = load_fragw(kvpack, nt * 4 + ks, lane);
      c = wmma16(a, b, c);
    }
    if (nt < 8) {                        // K output, token-major
      #pragma unroll
      for (int r = 0; r < 8; ++r) {
        int M = mt * 16 + r + ((lane >> 4) << 3);
        kbuf[M * 136 + n] = (_Float16)c[r];
      }
    } else {                             // V output, transposed (channel-major)
      int ch = (nt - 8) * 16 + (lane & 15);
      #pragma unroll
      for (int r = 0; r < 8; ++r) {
        int M = mt * 16 + r + ((lane >> 4) << 3);
        vT[ch * 72 + M] = (_Float16)c[r];
      }
    }
  }

  // ---- Phase F: q = xq16 @ q_w + q_b  (64x128 @ 128x128) ------------------
  #pragma unroll
  for (int rep = 0; rep < 4; ++rep) {
    int tile = wv + rep * 8;            // 0..31
    int mt = tile >> 3, nt = tile & 7;
    int n = nt * 16 + (lane & 15);
    float bv = q_b[n];
    v8f c;
    #pragma unroll
    for (int r = 0; r < 8; ++r) c[r] = bv;
    #pragma unroll
    for (int ks = 0; ks < 4; ++ks) {
      v16h a = load_frag16(xq16, 136, mt * 16, ks * 32, lane);
      v16h b = load_fragw(qpack, nt * 4 + ks, lane);
      c = wmma16(a, b, c);
    }
    #pragma unroll
    for (int r = 0; r < 8; ++r) {
      int M = mt * 16 + r + ((lane >> 4) << 3);
      qbuf[M * 136 + n] = (_Float16)c[r];
    }
  }
  __syncthreads();

  // ---- Phase G: sim = q @ k^T per head  (64x32 @ 32x64 per head) ----------
  #pragma unroll
  for (int rep = 0; rep < 8; ++rep) {
    int tile = wv + rep * 8;            // 0..63
    int h = tile >> 4, mt = (tile >> 2) & 3, nt = tile & 3;
    v16h a = load_frag16(qbuf, 136, mt * 16, h * 32, lane);
    v16h b = load_frag16(kbuf, 136, nt * 16, h * 32, lane);  // B lane = key token
    v8f c;
    #pragma unroll
    for (int r = 0; r < 8; ++r) c[r] = 0.f;
    c = wmma16(a, b, c);
    float* sb = sim + h * 64 * 68;
    #pragma unroll
    for (int r = 0; r < 8; ++r) {
      int q = mt * 16 + r + ((lane >> 4) << 3);
      sb[q * 68 + nt * 16 + (lane & 15)] = c[r];
    }
  }
  __syncthreads();

  // ---- Phase H: softmax with relative bias + shift mask -------------------
  {
    int h = tid >> 6, q = tid & 63;
    float* srow = sim + (h * 64 + q) * 68;
    int py = q >> 3, px = q & 7;
    bool lastY = (wy == GW - 1), lastX = (wx == GW - 1);
    const float scale = 0.17677669529663687f;   // 32^-0.5
    const float* rh = rpp + h * 225;            // [15][15]
    float mx = -3.4e38f;
    for (int k = 0; k < 64; ++k) {
      int ky = k >> 3, kx = k & 7;
      float s = srow[k] * scale + rh[(py - ky + 7) * 15 + (px - kx + 7)];
      bool masked = (lastY && ((py < SHF) != (ky < SHF))) ||
                    (lastX && ((px < SHF) != (kx < SHF)));
      if (masked) s = -3.4e38f;
      srow[k] = s;
      mx = fmaxf(mx, s);
    }
    float sum = 0.f;
    for (int k = 0; k < 64; ++k) {
      float e = __expf(srow[k] - mx);
      srow[k] = e;
      sum += e;
    }
    float inv = 1.f / sum;
    _Float16* prow = probs + (h * 64 + q) * 72;
    for (int k = 0; k < 64; ++k) prow[k] = (_Float16)(srow[k] * inv);
  }
  __syncthreads();

  // ---- Phase I: attnout = probs @ v per head  (64x64 @ 64x32) -------------
  #pragma unroll
  for (int rep = 0; rep < 4; ++rep) {
    int tile = wv + rep * 8;            // 0..31
    int h = tile >> 3, mt = (tile >> 1) & 3, nt = tile & 1;
    v8f c;
    #pragma unroll
    for (int r = 0; r < 8; ++r) c[r] = 0.f;
    #pragma unroll
    for (int ks = 0; ks < 2; ++ks) {
      v16h a = load_frag16(probs + h * 64 * 72, 72, mt * 16, ks * 32, lane);
      v16h b = load_frag16(vT + h * 32 * 72,    72, nt * 16, ks * 32, lane);
      c = wmma16(a, b, c);
    }
    int ch = h * 32 + nt * 16 + (lane & 15);
    #pragma unroll
    for (int r = 0; r < 8; ++r) {
      int M = mt * 16 + r + ((lane >> 4) << 3);
      aout[M * 136 + ch] = (_Float16)c[r];
    }
  }
  __syncthreads();

  // ---- Phase J: out = aout @ o_w + o_b, store un-shifted ------------------
  #pragma unroll
  for (int rep = 0; rep < 4; ++rep) {
    int tile = wv + rep * 8;            // 0..31
    int mt = tile >> 3, nt = tile & 7;
    int n = nt * 16 + (lane & 15);
    float bv = o_b[n];
    v8f c;
    #pragma unroll
    for (int r = 0; r < 8; ++r) c[r] = bv;
    #pragma unroll
    for (int ks = 0; ks < 4; ++ks) {
      v16h a = load_frag16(aout, 136, mt * 16, ks * 32, lane);
      v16h b = load_fragw(opack, nt * 4 + ks, lane);
      c = wmma16(a, b, c);
    }
    #pragma unroll
    for (int r = 0; r < 8; ++r) {
      int t = mt * 16 + r + ((lane >> 4) << 3);
      int i = t >> 3, j = t & 7;
      int row = (wy * 8 + i + SHF) & 255;   // un-shift == same mapping as load
      int col = (wx * 8 + j + SHF) & 255;
      out[(((size_t)bb * IMG + row) * IMG + col) * CCH + n] = c[r];
    }
  }
}

// ---------------------------------------------------------------------------
extern "C" void kernel_launch(void* const* d_in, const int* in_sizes, int n_in,
                              void* d_out, int out_size, void* d_ws, size_t ws_size,
                              hipStream_t stream) {
  const float* x    = (const float*)d_in[0];
  const float* rpp  = (const float*)d_in[1];
  const float* wi_w = (const float*)d_in[2];
  const float* wi_b = (const float*)d_in[3];
  const float* w1_w = (const float*)d_in[4];
  const float* w1_b = (const float*)d_in[5];
  const float* w2_w = (const float*)d_in[6];
  const float* w2_b = (const float*)d_in[7];
  const float* q_w  = (const float*)d_in[8];
  const float* q_b  = (const float*)d_in[9];
  const float* kv_w = (const float*)d_in[10];
  const float* kv_b = (const float*)d_in[11];
  const float* o_w  = (const float*)d_in[12];
  const float* o_b  = (const float*)d_in[13];

  _Float16* qp  = (_Float16*)d_ws;          // 16384 halves
  _Float16* kvp = qp + 16384;               // 32768 halves
  _Float16* op  = kvp + 32768;              // 16384 halves

  // Allow >64KB dynamic LDS (idempotent; safe during graph capture).
  (void)hipFuncSetAttribute((const void*)dgwa_main,
                            hipFuncAttributeMaxDynamicSharedMemorySize,
                            (int)SMEM_BYTES);

  dgwa_pack<<<256, 256, 0, stream>>>(q_w, kv_w, o_w, qp, kvp, op);
  dgwa_main<<<8 * GW * GW, 256, SMEM_BYTES, stream>>>(
      x, rpp, wi_w, wi_b, w1_w, w1_b, w2_w, w2_b,
      q_b, kv_b, o_b, qp, kvp, op, (float*)d_out);
}
